// ExtendedTransformer1_76467597738526
// MI455X (gfx1250) — compile-verified, matrix operand order checked by reading, not run on hardware
//
#include <hip/hip_runtime.h>

// ---------------- problem constants (from reference) ----------------
static constexpr int B_  = 2;
static constexpr int S_  = 1024;   // SD (decoder length)
static constexpr int D_  = 512;
static constexpr int H_  = 8;
static constexpr int DK_ = 64;
static constexpr int F_  = 2048;
static constexpr int V_  = 32000;
static constexpr int L_  = 4;
static constexpr int R_  = 2047;
static constexpr int M_  = B_ * S_;   // 2048 token rows

typedef unsigned short u16;
typedef unsigned int   u32;
typedef __attribute__((ext_vector_type(16))) __bf16 bf16x16;
typedef __attribute__((ext_vector_type(8)))  float  f32x8;

union Frag16 { bf16x16 v; uint4 q[2]; u16 s[16]; };
union Acc8   { f32x8 v; float f[8]; };
union Oct16  { uint4 q; u16 s[8]; };

__device__ __forceinline__ u16 f2bf(float x) {
  union { float f; u32 u; } c; c.f = x;
  u32 r = c.u + 0x7FFFu + ((c.u >> 16) & 1u);   // round-nearest-even
  return (u16)(r >> 16);
}

__device__ __forceinline__ f32x8 wmma_bf16(bf16x16 a, bf16x16 b, f32x8 c) {
  return __builtin_amdgcn_wmma_f32_16x16x32_bf16(false, a, false, b, (short)0, c,
                                                 false, false);
}

// ---------------- one-time f32 -> bf16 conversion (4 elems / thread) ----------
__global__ __launch_bounds__(256) void cvt_bf16_kernel(const float* __restrict__ in,
                                                       u16* __restrict__ out,
                                                       int n4) {   // n/4 quads
  int idx = blockIdx.x * 256 + threadIdx.x;
  if (idx >= n4) return;
  float4 v = *(const float4*)&in[(size_t)idx * 4];
  u32 lo = (u32)f2bf(v.x) | ((u32)f2bf(v.y) << 16);
  u32 hi = (u32)f2bf(v.z) | ((u32)f2bf(v.w) << 16);
  uint2* o = (uint2*)&out[(size_t)idx * 4];
  *o = make_uint2(lo, hi);
}

// ---------------- embedding gather ----------------
__global__ __launch_bounds__(256) void embed_kernel(const int* __restrict__ trg,
                                                    const float* __restrict__ emb,
                                                    float* __restrict__ yf,
                                                    u16* __restrict__ yb) {
  int idx = blockIdx.x * 256 + threadIdx.x;            // M_*D_ elements
  int t = idx / D_, d = idx - t * D_;
  int tok = trg[t];
  float v = emb[(size_t)tok * D_ + d];
  yf[idx] = v;
  yb[idx] = f2bf(v);
}

// ---------------- generic WMMA GEMM: C = A(bf16) @ W(bf16) + bias ------------
// Block: 128 threads = 4 waves; tile 64(M) x 128(N); K-step 32.
// Wave tile 32x64 = 2x4 WMMA accumulators -> 8 v_wmma per K-step.
// W is pre-converted bf16 [K][N]; staged transposed into LDS with pair-packed
// b32 stores. Epilogue fully compile-time (no branches in the store path).
template <bool RELU, bool OUTF, bool OUTB>
__global__ __launch_bounds__(128) void gemm_bf16(const u16* __restrict__ A,
                                                 const u16* __restrict__ Wb,
                                                 const float* __restrict__ bias,
                                                 float* __restrict__ outF,
                                                 u16* __restrict__ outB,
                                                 int M, int N, int K) {
  __shared__ __align__(16) u16 lA[64 * 32];    // [m][k], k contiguous
  __shared__ __align__(16) u16 lB[128 * 32];   // [n][k], k contiguous (W transposed)
  const int t    = threadIdx.x;
  const int lane = t & 31;
  const int wave = t >> 5;
  const int m0 = blockIdx.y * 64;
  const int n0 = blockIdx.x * 128;
  const int wm = (wave >> 1) * 32;
  const int wn = (wave & 1) * 64;
  const int half = (lane >> 4) & 1;
  const int ln   = lane & 15;

  Acc8 acc[2][4];
#pragma unroll
  for (int a = 0; a < 2; ++a)
#pragma unroll
    for (int b = 0; b < 4; ++b)
#pragma unroll
      for (int i = 0; i < 8; ++i) acc[a][b].f[i] = 0.f;

  for (int kk = 0; kk < K; kk += 32) {
    // stage A tile: 64x32 bf16 = 256 uint4 loads
#pragma unroll
    for (int g = 0; g < 2; ++g) {
      int e   = t + g * 128;            // [0,256)
      int row = e >> 2;
      int kb  = (e & 3) * 8;
      *(uint4*)&lA[row * 32 + kb] =
          *(const uint4*)&A[(size_t)(m0 + row) * K + kk + kb];
    }
    // stage W tile: 32x128 bf16, transpose to [n][k]; pack two k per b32 store
#pragma unroll
    for (int g = 0; g < 2; ++g) {
      int e  = t + g * 128;             // [0,256) octet-pair groups
      int kp = e >> 4;                  // 0..15 -> k = 2*kp
      int no = (e & 15) * 8;            // 0..120
      int k  = kp * 2;
      Oct16 r0, r1;
      r0.q = *(const uint4*)&Wb[(size_t)(kk + k) * N + n0 + no];
      r1.q = *(const uint4*)&Wb[(size_t)(kk + k + 1) * N + n0 + no];
#pragma unroll
      for (int j = 0; j < 8; ++j)
        *(u32*)&lB[(no + j) * 32 + k] = (u32)r0.s[j] | ((u32)r1.s[j] << 16);
    }
    // prefetch next K-tile while this one is consumed (global_prefetch_b8)
    if (kk + 32 < K) {
      __builtin_prefetch(&A[(size_t)(m0 + (t & 63)) * K + kk + 32], 0, 0);
      __builtin_prefetch(&Wb[(size_t)(kk + 32 + (t >> 2)) * N + n0 + (t & 3) * 32], 0, 0);
    }
    __syncthreads();

    Frag16 af[2], bf[4];
#pragma unroll
    for (int mi = 0; mi < 2; ++mi) {
      int r = wm + mi * 16 + ln;
      af[mi].q[0] = *(const uint4*)&lA[r * 32 + half * 8];
      af[mi].q[1] = *(const uint4*)&lA[r * 32 + half * 8 + 16];
    }
#pragma unroll
    for (int ni = 0; ni < 4; ++ni) {
      int r = wn + ni * 16 + ln;
      bf[ni].q[0] = *(const uint4*)&lB[r * 32 + half * 8];
      bf[ni].q[1] = *(const uint4*)&lB[r * 32 + half * 8 + 16];
    }
#pragma unroll
    for (int mi = 0; mi < 2; ++mi)
#pragma unroll
      for (int ni = 0; ni < 4; ++ni)
        acc[mi][ni].v = wmma_bf16(af[mi].v, bf[ni].v, acc[mi][ni].v);
    __syncthreads();
  }

#pragma unroll
  for (int mi = 0; mi < 2; ++mi)
#pragma unroll
    for (int ni = 0; ni < 4; ++ni) {
      int col  = n0 + wn + ni * 16 + ln;
      float bv = bias[col];
#pragma unroll
      for (int i = 0; i < 8; ++i) {
        int row = m0 + wm + mi * 16 + i + 8 * half;
        float v = acc[mi][ni].f[i] + bv;
        if (RELU) v = v > 0.f ? v : 0.f;
        size_t o = (size_t)row * N + col;
        if (OUTF) outF[o] = v;
        if (OUTB) outB[o] = f2bf(v);
      }
    }
}

// ---------------- attention scores: one 16x16 tile per wave ----------------
// s = (Q.K^T + Q.E[rid]^T)/8 + Bt[rid] - 1e9*max(causal, pad)
// E is pre-converted bf16 [R][64].
__global__ __launch_bounds__(32) void attn_scores(const u16* __restrict__ Q,
                                                  const u16* __restrict__ Km,
                                                  const u16* __restrict__ Eb,
                                                  const float* __restrict__ Bt,
                                                  const int* __restrict__ trg,
                                                  float* __restrict__ scores) {
  __shared__ __align__(16) float T[16 * 32];
  const int lane = threadIdx.x;
  const int half = (lane >> 4) & 1;
  const int ln   = lane & 15;
  const int kt = blockIdx.x, qt = blockIdx.y, z = blockIdx.z;
  const int b = z >> 3, h = z & 7;
  const int q0 = qt * 16, k0 = kt * 16;
  const int base = k0 - q0 + (S_ - 1) - 15;   // rid = base + (n - m + 15)

  // A fragments: Q rows q0..q0+15, DK=64 in two K=32 substeps
  Frag16 af[2], kf[2];
#pragma unroll
  for (int s = 0; s < 2; ++s) {
    int f0 = h * DK_ + s * 32 + half * 8;
    const u16* qp = Q + (size_t)(b * S_ + q0 + ln) * D_ + f0;
    af[s].q[0] = *(const uint4*)qp;
    af[s].q[1] = *(const uint4*)(qp + 16);
    const u16* kp = Km + (size_t)(b * S_ + k0 + ln) * D_ + f0;
    kf[s].q[0] = *(const uint4*)kp;
    kf[s].q[1] = *(const uint4*)(kp + 16);
  }

  Acc8 sA; Acc8 tA[2];
#pragma unroll
  for (int i = 0; i < 8; ++i) { sA.f[i] = 0.f; tA[0].f[i] = 0.f; tA[1].f[i] = 0.f; }
#pragma unroll
  for (int s = 0; s < 2; ++s) sA.v = wmma_bf16(af[s].v, kf[s].v, sA.v);

  // Rel term: T = Q_tile @ Eloc^T  (Eloc[j] = E[clip(base+j)], j in [0,32))
#pragma unroll
  for (int hN = 0; hN < 2; ++hN) {
    int j   = hN * 16 + ln;
    int rid = base + j; rid = rid < 0 ? 0 : (rid > R_ - 1 ? R_ - 1 : rid);
    const u16* er = Eb + (size_t)rid * DK_;
#pragma unroll
    for (int s = 0; s < 2; ++s) {
      int f0 = s * 32 + half * 8;
      Frag16 ef;
      ef.q[0] = *(const uint4*)(er + f0);
      ef.q[1] = *(const uint4*)(er + f0 + 16);
      tA[hN].v = wmma_bf16(af[s].v, ef.v, tA[hN].v);
    }
  }
  // spill T tile (16x32) to LDS for the diagonal gather
#pragma unroll
  for (int i = 0; i < 8; ++i) {
    int m = i + 8 * half;
    T[m * 32 + ln]      = tA[0].f[i];
    T[m * 32 + 16 + ln] = tA[1].f[i];
  }
  __syncthreads();

  const int tk = trg[b * S_ + k0 + ln];     // pad mask for this lane's column
#pragma unroll
  for (int i = 0; i < 8; ++i) {
    int m = i + 8 * half;
    int qq = q0 + m, kk = k0 + ln;
    int jj  = ln - m + 15;                  // 0..30
    int rid = base + jj; rid = rid < 0 ? 0 : (rid > R_ - 1 ? R_ - 1 : rid);
    float v = (sA.f[i] + T[m * 32 + jj]) * 0.125f + Bt[rid];
    if (kk > qq || tk == 0) v -= 1e9f;      // max(causal, pad) per reference
    scores[((size_t)z * S_ + qq) * S_ + kk] = v;
  }
}

// ---------------- row softmax -> bf16 probabilities ----------------
__global__ __launch_bounds__(256) void softmax_rows(const float* __restrict__ sc,
                                                    u16* __restrict__ P) {
  __shared__ float red[256];
  const int row = blockIdx.x;               // B*H*S rows
  const int t   = threadIdx.x;
  const float* r = sc + (size_t)row * S_;
  float x[4];
#pragma unroll
  for (int j = 0; j < 4; ++j) x[j] = r[t + j * 256];
  float mx = fmaxf(fmaxf(x[0], x[1]), fmaxf(x[2], x[3]));
  red[t] = mx; __syncthreads();
  for (int s = 128; s > 0; s >>= 1) { if (t < s) red[t] = fmaxf(red[t], red[t + s]); __syncthreads(); }
  mx = red[0]; __syncthreads();
  float e[4], sum = 0.f;
#pragma unroll
  for (int j = 0; j < 4; ++j) { e[j] = __expf(x[j] - mx); sum += e[j]; }
  red[t] = sum; __syncthreads();
  for (int s = 128; s > 0; s >>= 1) { if (t < s) red[t] += red[t + s]; __syncthreads(); }
  float inv = 1.f / red[0];
  u16* p = P + (size_t)row * S_;
#pragma unroll
  for (int j = 0; j < 4; ++j) p[t + j * 256] = f2bf(e[j] * inv);
}

// ---------------- V transpose: Vt[bh][d][t] = V[b*S+t][h*64+d] ----------------
__global__ __launch_bounds__(256) void vt_kernel(const u16* __restrict__ Vp,
                                                 u16* __restrict__ Vt) {
  int idx = blockIdx.x * 256 + threadIdx.x;   // B*H*DK*S
  int bh  = idx >> 16;
  int rem = idx & 0xFFFF;
  int d = rem >> 10, t = rem & 1023;
  int b = bh >> 3, h = bh & 7;
  Vt[idx] = Vp[(size_t)(b * S_ + t) * D_ + h * DK_ + d];
}

// ---------------- O = P @ V : 16x64 per wave, K=1024 ----------------
__global__ __launch_bounds__(32) void attn_av(const u16* __restrict__ P,
                                              const u16* __restrict__ Vt,
                                              u16* __restrict__ O) {
  const int lane = threadIdx.x;
  const int half = (lane >> 4) & 1;
  const int ln   = lane & 15;
  const int qt = blockIdx.x, z = blockIdx.y;
  const int b = z >> 3, h = z & 7;
  const int q0 = qt * 16;

  Acc8 acc[4];
#pragma unroll
  for (int n = 0; n < 4; ++n)
#pragma unroll
    for (int i = 0; i < 8; ++i) acc[n].f[i] = 0.f;

  for (int kk = 0; kk < S_; kk += 32) {
    Frag16 pa;
    const u16* pp = P + ((size_t)z * S_ + q0 + ln) * S_ + kk + half * 8;
    pa.q[0] = *(const uint4*)pp;
    pa.q[1] = *(const uint4*)(pp + 16);
#pragma unroll
    for (int ni = 0; ni < 4; ++ni) {
      Frag16 vf;
      const u16* vp = Vt + ((size_t)z * DK_ + ni * 16 + ln) * S_ + kk + half * 8;
      vf.q[0] = *(const uint4*)vp;
      vf.q[1] = *(const uint4*)(vp + 16);
      acc[ni].v = wmma_bf16(pa.v, vf.v, acc[ni].v);
    }
  }
#pragma unroll
  for (int ni = 0; ni < 4; ++ni)
#pragma unroll
    for (int i = 0; i < 8; ++i) {
      int m = i + 8 * half;
      O[(size_t)(b * S_ + q0 + m) * D_ + h * DK_ + ni * 16 + ln] = f2bf(acc[ni].f[i]);
    }
}

// ---------------- residual add + LayerNorm ----------------
__global__ __launch_bounds__(256) void add_layernorm(const float* __restrict__ x,
                                                     const float* __restrict__ a,
                                                     const float* __restrict__ g,
                                                     const float* __restrict__ bt,
                                                     float* __restrict__ outF,
                                                     u16* __restrict__ outB) {
  __shared__ float red[256];
  const int row = blockIdx.x;
  const int t   = threadIdx.x;
  size_t base = (size_t)row * D_;
  float v0 = x[base + t] + a[base + t];
  float v1 = x[base + t + 256] + a[base + t + 256];
  red[t] = v0 + v1; __syncthreads();
  for (int s = 128; s > 0; s >>= 1) { if (t < s) red[t] += red[t + s]; __syncthreads(); }
  float mean = red[0] * (1.f / D_); __syncthreads();
  float d0 = v0 - mean, d1 = v1 - mean;
  red[t] = d0 * d0 + d1 * d1; __syncthreads();
  for (int s = 128; s > 0; s >>= 1) { if (t < s) red[t] += red[t + s]; __syncthreads(); }
  float rstd = rsqrtf(red[0] * (1.f / D_) + 1e-6f);
  float o0 = g[t] * d0 * rstd + bt[t];
  float o1 = g[t + 256] * d1 * rstd + bt[t + 256];
  outF[base + t] = o0; outF[base + t + 256] = o1;
  if (outB) { outB[base + t] = f2bf(o0); outB[base + t + 256] = f2bf(o1); }
}

// ---------------- host orchestration ----------------
extern "C" void kernel_launch(void* const* d_in, const int* in_sizes, int n_in,
                              void* d_out, int out_size, void* d_ws, size_t ws_size,
                              hipStream_t stream) {
  (void)in_sizes; (void)n_in; (void)out_size; (void)ws_size;
  // Live inputs only (encoder + cross-attn are dead code in the reference).
  const int*   trg     = (const int*)d_in[1];
  const float* emb_dec = (const float*)d_in[6];
  const float* fc_W    = (const float*)d_in[7];
  const float* fc_b    = (const float*)d_in[8];
  const float* Wq = (const float*)d_in[17];
  const float* Wk = (const float*)d_in[18];
  const float* Wv = (const float*)d_in[19];
  const float* Wo = (const float*)d_in[20];
  const float* bq = (const float*)d_in[21];
  const float* bk = (const float*)d_in[22];
  const float* bv = (const float*)d_in[23];
  const float* bo = (const float*)d_in[24];
  const float* Et = (const float*)d_in[39];
  const float* Bt = (const float*)d_in[40];
  const float* W1 = (const float*)d_in[41];
  const float* b1 = (const float*)d_in[42];
  const float* W2 = (const float*)d_in[43];
  const float* b2 = (const float*)d_in[44];
  const float* ln1g = (const float*)d_in[49];
  const float* ln1b = (const float*)d_in[50];
  const float* ln2g = (const float*)d_in[51];
  const float* ln2b = (const float*)d_in[52];
  const float* ln3g = (const float*)d_in[53];
  const float* ln3b = (const float*)d_in[54];

  // workspace carve-out
  char* w = (char*)d_ws;
  auto carve = [&](size_t bytes) { char* p = w; w += (bytes + 255) & ~(size_t)255; return p; };
  float* y_f    = (float*)carve((size_t)M_ * D_ * 4);
  float* aout_f = (float*)carve((size_t)M_ * D_ * 4);
  float* out1_f = (float*)carve((size_t)M_ * D_ * 4);
  float* ffin_f = (float*)carve((size_t)M_ * D_ * 4);
  float* ff2_f  = (float*)carve((size_t)M_ * D_ * 4);
  u16* y_b    = (u16*)carve((size_t)M_ * D_ * 2);
  u16* q_b    = (u16*)carve((size_t)M_ * D_ * 2);
  u16* k_b    = (u16*)carve((size_t)M_ * D_ * 2);
  u16* v_b    = (u16*)carve((size_t)M_ * D_ * 2);
  u16* vt_b   = (u16*)carve((size_t)M_ * D_ * 2);
  u16* o_b    = (u16*)carve((size_t)M_ * D_ * 2);
  u16* ffin_b = (u16*)carve((size_t)M_ * D_ * 2);
  u16* h1_b   = (u16*)carve((size_t)M_ * F_ * 2);
  float* sc_f = (float*)carve((size_t)B_ * H_ * S_ * S_ * 4);
  u16*   p_b  = (u16*)carve((size_t)B_ * H_ * S_ * S_ * 2);
  // pre-converted bf16 weights (all layers) + E tables + fc
  u16* wq_b  = (u16*)carve((size_t)L_ * D_ * D_ * 2);
  u16* wk_b  = (u16*)carve((size_t)L_ * D_ * D_ * 2);
  u16* wv_b  = (u16*)carve((size_t)L_ * D_ * D_ * 2);
  u16* wo_b  = (u16*)carve((size_t)L_ * D_ * D_ * 2);
  u16* w1_b  = (u16*)carve((size_t)L_ * D_ * F_ * 2);
  u16* w2_b  = (u16*)carve((size_t)L_ * F_ * D_ * 2);
  u16* fcw_b = (u16*)carve((size_t)D_ * V_ * 2);
  u16* e_b   = (u16*)carve((size_t)L_ * R_ * DK_ * 2);

  // one-time weight conversions (amortized across all 32 M-tiles per GEMM)
  auto cvt = [&](const float* src, u16* dst, size_t n) {
    int n4 = (int)(n / 4);
    cvt_bf16_kernel<<<(n4 + 255) / 256, 256, 0, stream>>>(src, dst, n4);
  };
  cvt(Wq, wq_b, (size_t)L_ * D_ * D_);
  cvt(Wk, wk_b, (size_t)L_ * D_ * D_);
  cvt(Wv, wv_b, (size_t)L_ * D_ * D_);
  cvt(Wo, wo_b, (size_t)L_ * D_ * D_);
  cvt(W1, w1_b, (size_t)L_ * D_ * F_);
  cvt(W2, w2_b, (size_t)L_ * F_ * D_);
  cvt(fc_W, fcw_b, (size_t)D_ * V_);
  cvt(Et, e_b, (size_t)L_ * R_ * DK_);

  embed_kernel<<<(M_ * D_) / 256, 256, 0, stream>>>(trg, emb_dec, y_f, y_b);

  for (int l = 0; l < L_; ++l) {
    const u16* Wq_l = wq_b + (size_t)l * D_ * D_;
    const u16* Wk_l = wk_b + (size_t)l * D_ * D_;
    const u16* Wv_l = wv_b + (size_t)l * D_ * D_;
    const u16* Wo_l = wo_b + (size_t)l * D_ * D_;
    const float* bq_l = bq + (size_t)l * D_;
    const float* bk_l = bk + (size_t)l * D_;
    const float* bv_l = bv + (size_t)l * D_;
    const float* bo_l = bo + (size_t)l * D_;
    const u16*   E_l  = e_b + (size_t)l * R_ * DK_;
    const float* B_l  = Bt + (size_t)l * R_;
    const u16*   W1_l = w1_b + (size_t)l * D_ * F_;
    const float* b1_l = b1 + (size_t)l * F_;
    const u16*   W2_l = w2_b + (size_t)l * F_ * D_;
    const float* b2_l = b2 + (size_t)l * D_;

    dim3 gProj(D_ / 128, M_ / 64);   // N=512 tiles of 128
    gemm_bf16<false, false, true><<<gProj, 128, 0, stream>>>(y_b, Wq_l, bq_l, nullptr, q_b, M_, D_, D_);
    gemm_bf16<false, false, true><<<gProj, 128, 0, stream>>>(y_b, Wk_l, bk_l, nullptr, k_b, M_, D_, D_);
    gemm_bf16<false, false, true><<<gProj, 128, 0, stream>>>(y_b, Wv_l, bv_l, nullptr, v_b, M_, D_, D_);
    vt_kernel<<<(B_ * H_ * DK_ * S_) / 256, 256, 0, stream>>>(v_b, vt_b);

    attn_scores<<<dim3(S_ / 16, S_ / 16, B_ * H_), 32, 0, stream>>>(q_b, k_b, E_l, B_l, trg, sc_f);
    softmax_rows<<<B_ * H_ * S_, 256, 0, stream>>>(sc_f, p_b);
    attn_av<<<dim3(S_ / 16, B_ * H_), 32, 0, stream>>>(p_b, vt_b, o_b);
    gemm_bf16<false, true, false><<<gProj, 128, 0, stream>>>(o_b, Wo_l, bo_l, aout_f, nullptr, M_, D_, D_);

    add_layernorm<<<M_, 256, 0, stream>>>(y_f, aout_f, ln1g + l * D_, ln1b + l * D_, out1_f, nullptr);
    add_layernorm<<<M_, 256, 0, stream>>>(out1_f, aout_f, ln2g + l * D_, ln2b + l * D_, ffin_f, ffin_b);

    gemm_bf16<true, false, true><<<dim3(F_ / 128, M_ / 64), 128, 0, stream>>>(ffin_b, W1_l, b1_l, nullptr, h1_b, M_, F_, D_);
    gemm_bf16<false, true, false><<<dim3(D_ / 128, M_ / 64), 128, 0, stream>>>(h1_b, W2_l, b2_l, ff2_f, nullptr, M_, D_, F_);

    add_layernorm<<<M_, 256, 0, stream>>>(ffin_f, ff2_f, ln3g + l * D_, ln3b + l * D_, y_f, y_b);
  }

  // final projection straight into d_out (f32, [B*SD, V])
  gemm_bf16<false, true, false><<<dim3(V_ / 128, M_ / 64), 128, 0, stream>>>(
      y_b, fcw_b, fc_b, (float*)d_out, nullptr, M_, V_, D_);
}
